// UnnamedModel_75316546502661
// MI455X (gfx1250) — compile-verified, matrix operand order checked by reading, not run on hardware
//
#include <hip/hip_runtime.h>
#include <hip/hip_bf16.h>

// ---------------------------------------------------------------------------
// Types for CDNA5 WMMA
// ---------------------------------------------------------------------------
typedef __attribute__((ext_vector_type(16))) __bf16 v16bf;
typedef __attribute__((ext_vector_type(8)))  float  v8f;

union FragBF { v16bf v; uint4 q[2]; };

__device__ inline v8f v8f_zero() {
    v8f z;
    #pragma unroll
    for (int i = 0; i < 8; ++i) z[i] = 0.0f;
    return z;
}

#define LDT 40  // LDS row stride in bf16 elems (80B: 16B-aligned, conflict-free frags)

// ---------------------------------------------------------------------------
// Tiled GEMM:  C[M,N] = act(A[M,Kp] @ Wt[N,Kp]^T + bias)
//   A  : bf16 row-major, rows padded to mult of 128, Kp mult of 32 (zero pad)
//   Wt : bf16 "weights transposed" [N,Kp], N mult of 128
//   Cf : optional fp32 out [M,N];  Cb : optional bf16 out [M,N]
// WG tile 128x128, 8 waves (4M x 2N), wave tile 32x64 (2x4 WMMA accs).
// ---------------------------------------------------------------------------
__global__ __launch_bounds__(256)
void wmma_gemm_kernel(const __bf16* __restrict__ A, const __bf16* __restrict__ Wt,
                      const float* __restrict__ bias,
                      float* __restrict__ Cf, __bf16* __restrict__ Cb,
                      int M, int N, int Kp, int relu)
{
    __shared__ __align__(16) __bf16 sA[128 * LDT];
    __shared__ __align__(16) __bf16 sB[128 * LDT];

    const int tid  = threadIdx.x;
    const int lane = tid & 31;
    const int wid  = tid >> 5;
    const int wm   = wid & 3;   // 0..3 -> 32-row slab
    const int wn   = wid >> 2;  // 0..1 -> 64-col slab
    const int bm   = blockIdx.x * 128;
    const int bn   = blockIdx.y * 128;
    const int l15  = lane & 15;
    const int hi   = lane >> 4;

    // staging: thread moves 16 bf16 (32B) of one A row and one Wt row
    const int sr = tid >> 1;          // 0..127
    const int sh = (tid & 1) * 16;    // 0 or 16 (k sub-chunk)
    const __bf16* gA = A  + (size_t)(bm + sr) * Kp + sh;
    const __bf16* gB = Wt + (size_t)(bn + sr) * Kp + sh;
    __bf16* stA = &sA[sr * LDT + sh];
    __bf16* stB = &sB[sr * LDT + sh];

    v8f acc[2][4];
    #pragma unroll
    for (int i = 0; i < 2; ++i)
        #pragma unroll
        for (int j = 0; j < 4; ++j) acc[i][j] = v8f_zero();

    for (int k0 = 0; k0 < Kp; k0 += 32) {
        uint4 a0 = *(const uint4*)(gA + k0);
        uint4 a1 = *(const uint4*)(gA + k0 + 8);
        uint4 b0 = *(const uint4*)(gB + k0);
        uint4 b1 = *(const uint4*)(gB + k0 + 8);
        if (k0 + 128 < Kp) {  // prefetch 4 K-steps ahead, WGP-local (locality 3)
            __builtin_prefetch(gA + k0 + 128, 0, 3);
            __builtin_prefetch(gB + k0 + 128, 0, 3);
        }
        *(uint4*)(stA)     = a0;
        *(uint4*)(stA + 8) = a1;
        *(uint4*)(stB)     = b0;
        *(uint4*)(stB + 8) = b1;
        __syncthreads();

        // Fragments per CDNA5 16-bit 16x32 layout:
        // lane 0-15: elems 0..7 -> K=hi*8..+7 ; elems 8..15 -> K=16+hi*8..+7
        v16bf af[2], bf[4];
        #pragma unroll
        for (int mt = 0; mt < 2; ++mt) {
            const __bf16* p = &sA[(wm * 32 + mt * 16 + l15) * LDT];
            FragBF f;
            f.q[0] = *(const uint4*)(p + hi * 8);
            f.q[1] = *(const uint4*)(p + 16 + hi * 8);
            af[mt] = f.v;
        }
        #pragma unroll
        for (int nt = 0; nt < 4; ++nt) {
            const __bf16* p = &sB[(wn * 64 + nt * 16 + l15) * LDT];
            FragBF f;
            f.q[0] = *(const uint4*)(p + hi * 8);
            f.q[1] = *(const uint4*)(p + 16 + hi * 8);
            bf[nt] = f.v;
        }

        #pragma unroll
        for (int mt = 0; mt < 2; ++mt)
            #pragma unroll
            for (int nt = 0; nt < 4; ++nt)
                acc[mt][nt] = __builtin_amdgcn_wmma_f32_16x16x32_bf16(
                    false, af[mt], false, bf[nt], (short)0, acc[mt][nt],
                    false, false);
        __syncthreads();
    }

    // Epilogue: lane l, vgpr v -> M = v + (l>>4)*8, N = l&15
    #pragma unroll
    for (int nt = 0; nt < 4; ++nt) {
        int col = bn + wn * 64 + nt * 16 + l15;
        float bval = bias ? bias[col] : 0.0f;
        #pragma unroll
        for (int mt = 0; mt < 2; ++mt) {
            #pragma unroll
            for (int v = 0; v < 8; ++v) {
                int row = bm + wm * 32 + mt * 16 + hi * 8 + v;
                if (row >= M) continue;
                float val = acc[mt][nt][v] + bval;
                if (relu) val = fmaxf(val, 0.0f);
                if (Cf) Cf[(size_t)row * N + col] = val;
                if (Cb) Cb[(size_t)row * N + col] = (__bf16)val;
            }
        }
    }
}

// ---------------------------------------------------------------------------
// fp32 [M,K] -> bf16 [Mp,Kp] row-major, zero padded; 8 elems/thread, 16B stores
// ---------------------------------------------------------------------------
__global__ void cvt_a_kernel(const float* __restrict__ x, __bf16* __restrict__ y,
                             int M, int K, int Kp, int total8)
{
    int c = blockIdx.x * blockDim.x + threadIdx.x;
    if (c >= total8) return;
    int idx = c * 8;
    int r = idx / Kp, k = idx - r * Kp;
    __bf16 tmp[8];
    if (r < M) {
        const float* xp = x + (size_t)r * K;
        #pragma unroll
        for (int i = 0; i < 8; ++i) {
            int kk = k + i;
            tmp[i] = (__bf16)((kk < K) ? xp[kk] : 0.0f);
        }
    } else {
        #pragma unroll
        for (int i = 0; i < 8; ++i) tmp[i] = (__bf16)0.0f;
    }
    *(uint4*)(y + idx) = *(uint4*)tmp;
}

// ---------------------------------------------------------------------------
// fp32 W[K,N] -> bf16 Wt[N,Kp] (transpose + pad), 32x32 LDS tiles
// grid: (Kp/32, N/32)  (N mult of 32)
// ---------------------------------------------------------------------------
__global__ __launch_bounds__(256)
void cvt_w_kernel(const float* __restrict__ w, __bf16* __restrict__ wt,
                  int K, int N, int Kp)
{
    __shared__ __bf16 tile[32][33];
    const int k0 = blockIdx.x * 32, n0 = blockIdx.y * 32;
    const int t = threadIdx.x;
    #pragma unroll
    for (int i = 0; i < 4; ++i) {
        int idx = t + i * 256;          // 0..1023
        int kk = idx >> 5, nn = idx & 31;
        int gk = k0 + kk, gn = n0 + nn;
        float v = (gk < K && gn < N) ? w[(size_t)gk * N + gn] : 0.0f;
        tile[kk][nn] = (__bf16)v;
    }
    __syncthreads();
    int nl = t >> 3, kc = (t & 7) * 4;
    __bf16 tmp[4];
    #pragma unroll
    for (int i = 0; i < 4; ++i) tmp[i] = tile[kc + i][nl];
    *(uint2*)(wt + (size_t)(n0 + nl) * Kp + (k0 + kc)) = *(uint2*)tmp;
}

// ---------------------------------------------------------------------------
// Row dot: out[i] = dot(h[i,0:128], a)  (one wave32 per row)
// ---------------------------------------------------------------------------
__global__ void rowdot_kernel(const float* __restrict__ h, const float* __restrict__ a,
                              float* __restrict__ out, int n)
{
    int row  = blockIdx.x * 8 + (threadIdx.x >> 5);
    int lane = threadIdx.x & 31;
    if (row >= n) return;
    const float* hp = h + (size_t)row * 128;
    float s = 0.0f;
    #pragma unroll
    for (int j = 0; j < 4; ++j) s += hp[lane + j * 32] * a[lane + j * 32];
    #pragma unroll
    for (int off = 16; off > 0; off >>= 1) s += __shfl_down(s, off, 32);
    if (lane == 0) out[row] = s;
}

// ---------------------------------------------------------------------------
// Monotone float <-> uint encoding for atomicMax on floats
// ---------------------------------------------------------------------------
__device__ inline unsigned enc_f32(float f) {
    unsigned u = __float_as_uint(f);
    return (u & 0x80000000u) ? ~u : (u | 0x80000000u);
}
__device__ inline float dec_f32(unsigned u) {
    return (u & 0x80000000u) ? __uint_as_float(u & 0x7FFFFFFFu)
                             : __uint_as_float(~u);
}
__device__ inline float lrelu(float v) { return v > 0.0f ? v : 0.2f * v; }

__global__ void fill_u32_kernel(unsigned* p, unsigned v, int n) {
    int i = blockIdx.x * blockDim.x + threadIdx.x;
    if (i < n) p[i] = v;
}

__global__ void init_acc_kernel(float* acc, const float* b1, const float* b2, int total) {
    int i = blockIdx.x * blockDim.x + threadIdx.x;
    if (i >= total) return;
    int h = i & 127;
    acc[i] = b1[h] + b2[h];
}

__global__ void edge_max_kernel(const float* __restrict__ es, const float* __restrict__ ed,
                                const int* __restrict__ src, const int* __restrict__ dst,
                                unsigned* __restrict__ menc, int E)
{
    int e = blockIdx.x * blockDim.x + threadIdx.x;
    if (e >= E) return;
    float v = lrelu(es[src[e]] + ed[dst[e]]);
    atomicMax(&menc[dst[e]], enc_f32(v));
}

__global__ void edge_den_kernel(const float* __restrict__ es, const float* __restrict__ ed,
                                const int* __restrict__ src, const int* __restrict__ dst,
                                const unsigned* __restrict__ menc, float* __restrict__ den, int E)
{
    int e = blockIdx.x * blockDim.x + threadIdx.x;
    if (e >= E) return;
    int d = dst[e];
    float v = lrelu(es[src[e]] + ed[d]);
    unsafeAtomicAdd(&den[d], expf(v - dec_f32(menc[d])));
}

// One wave32 per edge; 128-wide weighted gather -> scatter-add
__global__ void edge_aggr_kernel(const float* __restrict__ es, const float* __restrict__ ed,
                                 const int* __restrict__ src, const int* __restrict__ dst,
                                 const unsigned* __restrict__ menc, const float* __restrict__ den,
                                 const float* __restrict__ hs, float* __restrict__ acc, int E)
{
    int e    = blockIdx.x * 8 + (threadIdx.x >> 5);
    int lane = threadIdx.x & 31;
    if (e >= E) return;
    int s = src[e], d = dst[e];
    float v = lrelu(es[s] + ed[d]);
    float alpha = expf(v - dec_f32(menc[d])) / (den[d] + 1e-16f);
    const float* hp = hs + (size_t)s * 128;
    float*       ap = acc + (size_t)d * 128;
    #pragma unroll
    for (int j = 0; j < 4; ++j) {
        int c = lane + j * 32;
        unsafeAtomicAdd(&ap[c], alpha * hp[c]);
    }
}

__global__ void relu_copy_kernel(const float* __restrict__ in, float* __restrict__ out, int n) {
    int i = blockIdx.x * blockDim.x + threadIdx.x;
    if (i < n) out[i] = fmaxf(in[i], 0.0f);
}

// L2-normalize each row of x [rows, cols] -> bf16 y [rows, Kp] (zero-padded)
__global__ void rownorm_bf16_kernel(const float* __restrict__ x, __bf16* __restrict__ y,
                                    int cols, int Kp)
{
    __shared__ float red[256];
    int row = blockIdx.x;
    const float* xp = x + (size_t)row * cols;
    float s = 0.0f;
    for (int j = threadIdx.x; j < cols; j += 256) { float v = xp[j]; s += v * v; }
    red[threadIdx.x] = s;
    __syncthreads();
    for (int o = 128; o > 0; o >>= 1) {
        if (threadIdx.x < o) red[threadIdx.x] += red[threadIdx.x + o];
        __syncthreads();
    }
    float inv = 1.0f / fmaxf(sqrtf(red[0]), 1e-12f);
    __bf16* yp = y + (size_t)row * Kp;
    for (int j = threadIdx.x; j < Kp; j += 256)
        yp[j] = (__bf16)((j < cols) ? xp[j] * inv : 0.0f);
}

// hidden[b, 0:512] = [drug_out[d1[b]], drug_out[d2[b]], cell[b]]
__global__ void concat_kernel(const float* __restrict__ drug_out, const float* __restrict__ cell,
                              const int* __restrict__ d1, const int* __restrict__ d2,
                              float* __restrict__ hid, int total)
{
    int i = blockIdx.x * blockDim.x + threadIdx.x;
    if (i >= total) return;
    int b = i >> 9, c = i & 511;
    float v;
    if (c < 128)      v = drug_out[(size_t)d1[b] * 128 + c];
    else if (c < 256) v = drug_out[(size_t)d2[b] * 128 + (c - 128)];
    else              v = cell[(size_t)b * 256 + (c - 256)];
    hid[i] = v;
}

// logits[b,0:2] = hfin[b,0:128] @ W[128,2] + b (one wave32 per row)
__global__ void cls_kernel(const float* __restrict__ hfin, const float* __restrict__ W,
                           const float* __restrict__ b, float* __restrict__ logits, int B)
{
    int row  = blockIdx.x * 8 + (threadIdx.x >> 5);
    int lane = threadIdx.x & 31;
    if (row >= B) return;
    const float* hp = hfin + (size_t)row * 128;
    float s0 = 0.0f, s1 = 0.0f;
    #pragma unroll
    for (int j = 0; j < 4; ++j) {
        int c = lane + j * 32;
        float v = hp[c];
        s0 += v * W[c * 2 + 0];
        s1 += v * W[c * 2 + 1];
    }
    #pragma unroll
    for (int off = 16; off > 0; off >>= 1) {
        s0 += __shfl_down(s0, off, 32);
        s1 += __shfl_down(s1, off, 32);
    }
    if (lane == 0) { logits[row * 2 + 0] = s0 + b[0]; logits[row * 2 + 1] = s1 + b[1]; }
}

// ---------------------------------------------------------------------------
// Host side
// ---------------------------------------------------------------------------
static inline void launch_gemm(const __bf16* A, const __bf16* Wt, const float* bias,
                               float* Cf, __bf16* Cb, int M, int Mp, int N, int Kp,
                               int relu, hipStream_t s)
{
    dim3 g(Mp / 128, N / 128), b(256);
    wmma_gemm_kernel<<<g, b, 0, s>>>(A, Wt, bias, Cf, Cb, M, N, Kp, relu);
}

static inline void launch_cvt_a(const float* x, __bf16* y, int M, int K, int Mp, int Kp,
                                hipStream_t s)
{
    int total8 = (Mp * Kp) / 8;
    cvt_a_kernel<<<dim3((total8 + 255) / 256), dim3(256), 0, s>>>(x, y, M, K, Kp, total8);
}

static inline void launch_cvt_w(const float* w, __bf16* wt, int K, int N, int Kp,
                                hipStream_t s)
{
    cvt_w_kernel<<<dim3(Kp / 32, N / 32), dim3(256), 0, s>>>(w, wt, K, N, Kp);
}

extern "C" void kernel_launch(void* const* d_in, const int* in_sizes, int n_in,
                              void* d_out, int out_size, void* d_ws, size_t ws_size,
                              hipStream_t stream)
{
    const int ND = 10000, NT = 10000, B = 8192;
    const int DF = 256, TF = 256, H = 128, CF = 954, P1 = 2048, P2 = 512;
    const int NDp = 10112, NTp = 10112;        // rounded to 128
    const int CFp = 960;                       // 954 -> mult of 32

    const float* x_drug   = (const float*)d_in[0];
    const float* x_target = (const float*)d_in[1];
    const float* cellf    = (const float*)d_in[2];
    const float* dd_W  = (const float*)d_in[3];
    const float* dd_as = (const float*)d_in[4];
    const float* dd_ad = (const float*)d_in[5];
    const float* dd_b  = (const float*)d_in[6];
    const float* dt_Ws = (const float*)d_in[7];
    const float* dt_Wd = (const float*)d_in[8];
    const float* dt_as = (const float*)d_in[9];
    const float* dt_ad = (const float*)d_in[10];
    const float* dt_b  = (const float*)d_in[11];
    const float* td_Ws = (const float*)d_in[12];
    const float* td_Wd = (const float*)d_in[13];
    const float* td_as = (const float*)d_in[14];
    const float* td_ad = (const float*)d_in[15];
    const float* td_b  = (const float*)d_in[16];
    const float* tt_W  = (const float*)d_in[17];
    const float* tt_as = (const float*)d_in[18];
    const float* tt_ad = (const float*)d_in[19];
    const float* tt_b  = (const float*)d_in[20];
    const float* r1_W1 = (const float*)d_in[21];
    const float* r1_b1 = (const float*)d_in[22];
    const float* r1_W2 = (const float*)d_in[23];
    const float* r1_b2 = (const float*)d_in[24];
    const float* r1_W3 = (const float*)d_in[25];
    const float* r1_b3 = (const float*)d_in[26];
    const float* r2_W1 = (const float*)d_in[27];
    const float* r2_b1 = (const float*)d_in[28];
    const float* r2_W2 = (const float*)d_in[29];
    const float* r2_b2 = (const float*)d_in[30];
    const float* r2_W3 = (const float*)d_in[31];
    const float* r2_b3 = (const float*)d_in[32];
    const float* cls_W = (const float*)d_in[33];
    const float* cls_b = (const float*)d_in[34];
    const int* e_dd_src = (const int*)d_in[35];
    const int* e_dd_dst = (const int*)d_in[36];
    const int* e_dt_src = (const int*)d_in[37];
    const int* e_dt_dst = (const int*)d_in[38];
    const int* e_td_src = (const int*)d_in[39];
    const int* e_td_dst = (const int*)d_in[40];
    const int* e_tt_src = (const int*)d_in[41];
    const int* e_tt_dst = (const int*)d_in[42];
    const int* drug1_id = (const int*)d_in[43];
    const int* drug2_id = (const int*)d_in[44];
    const int  E = in_sizes[35];

    float* out_logits = (float*)d_out;                       // [B,2]
    float* out_drug   = (float*)d_out + (size_t)B * 2;       // [ND,128]
    float* out_target = out_drug + (size_t)ND * H;           // [NT,128]

    // ---------------- workspace bump allocator (256B aligned) ----------------
    char*  base = (char*)d_ws;
    size_t off  = 0;
    auto alloc = [&](size_t bytes) -> void* {
        void* p = base + off;
        off = (off + bytes + 255) & ~(size_t)255;
        return p;
    };
    // fp32 buffers
    float* h_dd   = (float*)alloc((size_t)ND * H * 4);
    float* h_dt_s = (float*)alloc((size_t)ND * H * 4);
    float* h_dt_d = (float*)alloc((size_t)NT * H * 4);
    float* h_td_s = (float*)alloc((size_t)NT * H * 4);
    float* h_td_d = (float*)alloc((size_t)ND * H * 4);
    float* h_tt   = (float*)alloc((size_t)NT * H * 4);
    float* es_dd = (float*)alloc(ND * 4);
    float* ed_dd = (float*)alloc(ND * 4);
    float* es_dt = (float*)alloc(ND * 4);
    float* ed_dt = (float*)alloc(NT * 4);
    float* es_td = (float*)alloc(NT * 4);
    float* ed_td = (float*)alloc(ND * 4);
    float* es_tt = (float*)alloc(NT * 4);
    float* ed_tt = (float*)alloc(NT * 4);
    unsigned* m_enc = (unsigned*)alloc(10000 * 4);
    float* den      = (float*)alloc(10000 * 4);
    float* acc_drug   = (float*)alloc((size_t)ND * H * 4);
    float* acc_target = (float*)alloc((size_t)NT * H * 4);
    float* CELL = (float*)alloc((size_t)B * 2 * H * 4);
    float* HID  = (float*)alloc((size_t)B * 4 * H * 4);
    float* T3   = (float*)alloc((size_t)B * H * 4);
    // bf16 buffers
    __bf16* xd_bf  = (__bf16*)alloc((size_t)NDp * DF * 2);
    __bf16* xt_bf  = (__bf16*)alloc((size_t)NTp * TF * 2);
    __bf16* ddWt   = (__bf16*)alloc((size_t)H * DF * 2);
    __bf16* dtWst  = (__bf16*)alloc((size_t)H * DF * 2);
    __bf16* dtWdt  = (__bf16*)alloc((size_t)H * TF * 2);
    __bf16* tdWst  = (__bf16*)alloc((size_t)H * TF * 2);
    __bf16* tdWdt  = (__bf16*)alloc((size_t)H * DF * 2);
    __bf16* ttWt   = (__bf16*)alloc((size_t)H * TF * 2);
    __bf16* r1W1t  = (__bf16*)alloc((size_t)P1 * CFp * 2);
    __bf16* r1W2t  = (__bf16*)alloc((size_t)P2 * P1 * 2);
    __bf16* r1W3t  = (__bf16*)alloc((size_t)(2 * H) * P2 * 2);
    __bf16* r2W1t  = (__bf16*)alloc((size_t)P1 * (4 * H) * 2);
    __bf16* r2W2t  = (__bf16*)alloc((size_t)P2 * P1 * 2);
    __bf16* r2W3t  = (__bf16*)alloc((size_t)H * P2 * 2);
    __bf16* XNbf   = (__bf16*)alloc((size_t)B * CFp * 2);
    __bf16* T1bf   = (__bf16*)alloc((size_t)B * P1 * 2);
    __bf16* T2bf   = (__bf16*)alloc((size_t)B * P2 * 2);
    __bf16* HIDbf  = (__bf16*)alloc((size_t)B * 4 * H * 2);

    dim3 b256(256);

    // ---------------- 0) precision conversion (once) ----------------
    launch_cvt_a(x_drug,   xd_bf, ND, DF, NDp, DF, stream);
    launch_cvt_a(x_target, xt_bf, NT, TF, NTp, TF, stream);
    launch_cvt_w(dd_W,  ddWt,  DF, H, DF, stream);
    launch_cvt_w(dt_Ws, dtWst, DF, H, DF, stream);
    launch_cvt_w(dt_Wd, dtWdt, TF, H, TF, stream);
    launch_cvt_w(td_Ws, tdWst, TF, H, TF, stream);
    launch_cvt_w(td_Wd, tdWdt, DF, H, DF, stream);
    launch_cvt_w(tt_W,  ttWt,  TF, H, TF, stream);
    launch_cvt_w(r1_W1, r1W1t, CF, P1, CFp, stream);
    launch_cvt_w(r1_W2, r1W2t, P1, P2, P1,  stream);
    launch_cvt_w(r1_W3, r1W3t, P2, 2 * H, P2, stream);
    launch_cvt_w(r2_W1, r2W1t, 4 * H, P1, 4 * H, stream);
    launch_cvt_w(r2_W2, r2W2t, P1, P2, P1,  stream);
    launch_cvt_w(r2_W3, r2W3t, P2, H, P2,  stream);

    // ---------------- 1) node feature transforms (WMMA GEMMs) ----------------
    launch_gemm(xd_bf, ddWt,  nullptr, h_dd,   nullptr, ND, NDp, H, DF, 0, stream);
    launch_gemm(xd_bf, dtWst, nullptr, h_dt_s, nullptr, ND, NDp, H, DF, 0, stream);
    launch_gemm(xt_bf, dtWdt, nullptr, h_dt_d, nullptr, NT, NTp, H, TF, 0, stream);
    launch_gemm(xt_bf, tdWst, nullptr, h_td_s, nullptr, NT, NTp, H, TF, 0, stream);
    launch_gemm(xd_bf, tdWdt, nullptr, h_td_d, nullptr, ND, NDp, H, DF, 0, stream);
    launch_gemm(xt_bf, ttWt,  nullptr, h_tt,   nullptr, NT, NTp, H, TF, 0, stream);

    // ---------------- 2) attention logits ----------------
    rowdot_kernel<<<dim3((ND + 7) / 8), b256, 0, stream>>>(h_dd,   dd_as, es_dd, ND);
    rowdot_kernel<<<dim3((ND + 7) / 8), b256, 0, stream>>>(h_dd,   dd_ad, ed_dd, ND);
    rowdot_kernel<<<dim3((ND + 7) / 8), b256, 0, stream>>>(h_dt_s, dt_as, es_dt, ND);
    rowdot_kernel<<<dim3((NT + 7) / 8), b256, 0, stream>>>(h_dt_d, dt_ad, ed_dt, NT);
    rowdot_kernel<<<dim3((NT + 7) / 8), b256, 0, stream>>>(h_td_s, td_as, es_td, NT);
    rowdot_kernel<<<dim3((ND + 7) / 8), b256, 0, stream>>>(h_td_d, td_ad, ed_td, ND);
    rowdot_kernel<<<dim3((NT + 7) / 8), b256, 0, stream>>>(h_tt,   tt_as, es_tt, NT);
    rowdot_kernel<<<dim3((NT + 7) / 8), b256, 0, stream>>>(h_tt,   tt_ad, ed_tt, NT);

    // ---------------- 3) init accumulators with summed biases ----------------
    init_acc_kernel<<<dim3((ND * H + 255) / 256), b256, 0, stream>>>(acc_drug,   dd_b, td_b, ND * H);
    init_acc_kernel<<<dim3((NT * H + 255) / 256), b256, 0, stream>>>(acc_target, dt_b, tt_b, NT * H);

    // ---------------- 4) edge softmax + aggregate (sequential per type) -----
    const unsigned ENC_NEG_INF = 0x007FFFFFu;  // enc(-inf)
    struct Etype {
        const float* es; const float* ed; const int* src; const int* dst;
        const float* hs; float* acc; int n_dst;
    } types[4] = {
        { es_dd, ed_dd, e_dd_src, e_dd_dst, h_dd,   acc_drug,   ND },
        { es_td, ed_td, e_td_src, e_td_dst, h_td_s, acc_drug,   ND },
        { es_dt, ed_dt, e_dt_src, e_dt_dst, h_dt_s, acc_target, NT },
        { es_tt, ed_tt, e_tt_src, e_tt_dst, h_tt,   acc_target, NT },
    };
    for (int t = 0; t < 4; ++t) {
        const Etype& et = types[t];
        fill_u32_kernel<<<dim3((et.n_dst + 255) / 256), b256, 0, stream>>>(m_enc, ENC_NEG_INF, et.n_dst);
        fill_u32_kernel<<<dim3((et.n_dst + 255) / 256), b256, 0, stream>>>((unsigned*)den, 0u, et.n_dst);
        edge_max_kernel <<<dim3((E + 255) / 256), b256, 0, stream>>>(et.es, et.ed, et.src, et.dst, m_enc, E);
        edge_den_kernel <<<dim3((E + 255) / 256), b256, 0, stream>>>(et.es, et.ed, et.src, et.dst, m_enc, den, E);
        edge_aggr_kernel<<<dim3((E + 7) / 8),     b256, 0, stream>>>(et.es, et.ed, et.src, et.dst, m_enc, den,
                                                                     et.hs, et.acc, E);
    }

    // ---------------- 5) finalize GAT outputs ----------------
    relu_copy_kernel<<<dim3((ND * H + 255) / 256), b256, 0, stream>>>(acc_drug,   out_drug,   ND * H);
    relu_copy_kernel<<<dim3((NT * H + 255) / 256), b256, 0, stream>>>(acc_target, out_target, NT * H);

    // ---------------- 6) reduction MLP 1 (cell) ----------------
    rownorm_bf16_kernel<<<dim3(B), b256, 0, stream>>>(cellf, XNbf, CF, CFp);
    launch_gemm(XNbf, r1W1t, r1_b1, nullptr, T1bf, B, B, P1,    CFp, 1, stream);
    launch_gemm(T1bf, r1W2t, r1_b2, nullptr, T2bf, B, B, P2,    P1,  1, stream);
    launch_gemm(T2bf, r1W3t, r1_b3, CELL,    nullptr, B, B, 2 * H, P2, 1, stream);

    // ---------------- 7) concat + normalize + reduction MLP 2 ----------------
    concat_kernel<<<dim3((B * 4 * H + 255) / 256), b256, 0, stream>>>(out_drug, CELL, drug1_id, drug2_id,
                                                                      HID, B * 4 * H);
    rownorm_bf16_kernel<<<dim3(B), b256, 0, stream>>>(HID, HIDbf, 4 * H, 4 * H);
    launch_gemm(HIDbf, r2W1t, r2_b1, nullptr, T1bf, B, B, P1, 4 * H, 1, stream);
    launch_gemm(T1bf,  r2W2t, r2_b2, nullptr, T2bf, B, B, P2, P1,    1, stream);
    launch_gemm(T2bf,  r2W3t, r2_b3, T3, nullptr,   B, B, H,  P2,    1, stream);

    // ---------------- 8) classifier ----------------
    cls_kernel<<<dim3((B + 7) / 8), b256, 0, stream>>>(T3, cls_W, cls_b, out_logits, B);
}
